// SpectralGraphConv_48318382080441
// MI455X (gfx1250) — matrix operand order repaired and checked by reading, not compiled, not run on hardware
//
#include <hip/hip_runtime.h>
#include <cstdint>

// Problem constants (match reference):  B=8, N=1024, C_IN=C_OUT=64, K=4
#define NB   8
#define NN   1024
#define NC   64
#define NK   4

typedef __attribute__((ext_vector_type(16))) _Float16 v16h;
typedef __attribute__((ext_vector_type(8)))  _Float16 h8;
typedef __attribute__((ext_vector_type(4)))  _Float16 h4;
typedef __attribute__((ext_vector_type(8)))  float    v8f;

union Frag16 { v16h v; h8 h[2]; };

// Build a 16-half fragment from two contiguous 16-byte runs.
__device__ __forceinline__ v16h load_frag_split(const _Float16* p0, const _Float16* p1) {
    Frag16 f;
    f.h[0] = *(const h8*)p0;
    f.h[1] = *(const h8*)p1;
    return f.v;
}

// CDNA5 async global->LDS copy (16B per lane), tracked by ASYNCcnt.
__device__ __forceinline__ void async_b128(void* lds_ptr, const void* gptr) {
    uint32_t lds_off = (uint32_t)(uintptr_t)lds_ptr;
    asm volatile("global_load_async_to_lds_b128 %0, %1, off"
                 :: "v"(lds_off), "v"(gptr) : "memory");
}
__device__ __forceinline__ void wait_asynccnt0() {
    asm volatile("s_wait_asynccnt 0x0" ::: "memory");
}

// ---------------------------------------------------------------------------
// fp32 -> f16 bulk conversion (vectorized: float4 in, 4 halves out)
// ---------------------------------------------------------------------------
__global__ __launch_bounds__(256)
void cvt_f32_to_f16(const float* __restrict__ src, _Float16* __restrict__ dst, int n4) {
    int i = blockIdx.x * blockDim.x + threadIdx.x;
    if (i < n4) {
        float4 v = ((const float4*)src)[i];
        h4 o = { (_Float16)v.x, (_Float16)v.y, (_Float16)v.z, (_Float16)v.w };
        ((h4*)dst)[i] = o;
    }
}

// x [B][N][C] fp32 -> xh row-major f16 AND xhT [B][C][N] f16
__global__ __launch_bounds__(256)
void cvt_x_dual(const float* __restrict__ x, _Float16* __restrict__ xh,
                _Float16* __restrict__ xhT) {
    int t = blockIdx.x * blockDim.x + threadIdx.x;      // 0 .. 524287
    int c = t & 63;
    int n = (t >> 6) & 1023;
    int b = t >> 16;
    _Float16 v = (_Float16)x[t];
    xh[t] = v;
    xhT[((size_t)b * NC + c) * NN + n] = v;
}

// theta [K][C_IN][C_OUT] fp32  ->  thetaT [C_OUT][K*C_IN] f16
__global__ __launch_bounds__(256)
void pack_theta(const float* __restrict__ th, _Float16* __restrict__ thT) {
    int t = blockIdx.x * blockDim.x + threadIdx.x;      // 0 .. 16383
    int co = t & 63;
    int ci = (t >> 6) & 63;
    int k  = t >> 12;
    thT[co * (NK * NC) + k * NC + ci] = (_Float16)th[t];
}

// ---------------------------------------------------------------------------
// GEMM helpers
// ---------------------------------------------------------------------------
__device__ __forceinline__ void stage_chunk(_Float16 (*Asb)[64], _Float16 (*Bsb)[64],
                                            const _Float16* gA, const _Float16* gB,
                                            int k0, int sr, int so) {
    async_b128(&Asb[sr][so],     gA + k0);
    async_b128(&Asb[sr][so + 8], gA + k0 + 8);
    async_b128(&Bsb[sr][so],     gB + k0);
    async_b128(&Bsb[sr][so + 8], gB + k0 + 8);
}

// One 64-K chunk: 4 WMMAs across 4 independent accumulation chains.
__device__ __forceinline__ void compute_chunk(const _Float16 (*Asb)[64],
                                              const _Float16 (*Bsb)[64],
                                              int fr, int fkh, int fkg, int fc,
                                              v8f& acc0a, v8f& acc1a,
                                              v8f& acc0b, v8f& acc1b) {
    v16h a0 = load_frag_split(&Asb[fr][fkg],      &Asb[fr][fkg + 8]);
    v16h p0 = *(const v16h*)&Bsb[fc][fkg];
    v16h q0 = *(const v16h*)&Bsb[fc + 16][fkg];
    v16h a1 = load_frag_split(&Asb[fr][32 + fkg], &Asb[fr][40 + fkg]);
    v16h p1 = *(const v16h*)&Bsb[fc][32 + fkg];
    v16h q1 = *(const v16h*)&Bsb[fc + 16][32 + fkg];
    acc0a = __builtin_amdgcn_wmma_f32_16x16x32_f16(false, a0, false, p0, (short)0, acc0a, false, false);
    acc1a = __builtin_amdgcn_wmma_f32_16x16x32_f16(false, a0, false, q0, (short)0, acc1a, false, false);
    acc0b = __builtin_amdgcn_wmma_f32_16x16x32_f16(false, a1, false, p1, (short)0, acc0b, false, false);
    acc1b = __builtin_amdgcn_wmma_f32_16x16x32_f16(false, a1, false, q1, (short)0, acc1b, false, false);
}

// ---------------------------------------------------------------------------
// Znext = alpha * (L @ Zcur) + beta * Zprev      (per batch)
//   Lh:   f16 [NB][NN][NN] row-major
//   ZhT:  f16 [NB][NC][NN] column-major z_curr  (B operand, K-contiguous)
//   Zprev:f32 [NB][NN][NC]                      (ignored when beta == 0)
// Outputs: Znext fp32, ZnextH row-major f16, ZnextHT column-major f16.
// Block: 256 threads (8 waves), tile 64x64, K chunk 64, double-buffered
// async global->LDS staging, manual 2x unroll with static buffer indices.
// ---------------------------------------------------------------------------
__global__ __launch_bounds__(256)
void cheb_gemmL(const _Float16* __restrict__ Lh,
                const _Float16* __restrict__ ZhT,
                const float*    __restrict__ Zprev,
                float*          __restrict__ Znext,
                _Float16*       __restrict__ ZnextH,
                _Float16*       __restrict__ ZnextHT,
                float alpha, float beta) {
    __shared__ __align__(128) _Float16 As[2][64][64];
    __shared__ __align__(128) _Float16 Bs[2][64][64];

    const int b    = blockIdx.y;
    const int row0 = blockIdx.x * 64;
    const int t    = threadIdx.x;
    const int w    = t >> 5;
    const int l    = t & 31;
    const int wr   = w >> 1;     // 0..3 -> row group
    const int wc   = w & 1;      // 0..1 -> col group

    const _Float16* Lb  = Lh  + (size_t)b * NN * NN;
    const _Float16* ZbT = ZhT + (size_t)b * NC * NN;

    const int sr = t >> 2;
    const int so = (t & 3) * 16;                     // half offset 0,16,32,48
    const _Float16* gA = Lb  + (size_t)(row0 + sr) * NN + so;
    const _Float16* gB = ZbT + (size_t)sr * NN + so;

    const int fr  = wr * 16 + (l & 15);
    const int fkh = (l >> 4) * 8;
    const int fkg = (l >> 4) * 16;
    const int fc  = wc * 32 + (l & 15);

    v8f acc0a = {}, acc1a = {}, acc0b = {}, acc1b = {};

    // prologue: chunk 0 -> buffer 0
    stage_chunk(As[0], Bs[0], gA, gB, 0, sr, so);
    wait_asynccnt0();
    __syncthreads();

    const int NCHUNK = NN / 64;                      // 16 (even)
    for (int j = 0; j < NCHUNK; j += 2) {
        // chunk j+1 -> buffer 1 while computing buffer 0
        stage_chunk(As[1], Bs[1], gA, gB, (j + 1) * 64, sr, so);
        compute_chunk(As[0], Bs[0], fr, fkh, fkg, fc, acc0a, acc1a, acc0b, acc1b);
        wait_asynccnt0();
        __syncthreads();

        // chunk j+2 -> buffer 0 while computing buffer 1
        if (j + 2 < NCHUNK)
            stage_chunk(As[0], Bs[0], gA, gB, (j + 2) * 64, sr, so);
        compute_chunk(As[1], Bs[1], fr, fkh, fkg, fc, acc0a, acc1a, acc0b, acc1b);
        wait_asynccnt0();
        __syncthreads();
    }

    // C/D layout: VGPR i -> row = 8*(l>>4) + i, col = l&15
    const int orow = row0 + wr * 16 + (l >> 4) * 8;
    const int ocol = wc * 32 + (l & 15);
    float*       Zn   = Znext   + (size_t)b * NN * NC;
    _Float16*    ZnH  = ZnextH  + (size_t)b * NN * NC;
    _Float16*    ZnHT = ZnextHT + (size_t)b * NC * NN;
    const float* Zp   = Zprev   + (size_t)b * NN * NC;

    #pragma unroll
    for (int i = 0; i < 8; ++i) {
        const int rr = orow + i;
        float v0 = alpha * (acc0a[i] + acc0b[i]);
        float v1 = alpha * (acc1a[i] + acc1b[i]);
        if (beta != 0.0f) {
            v0 += beta * Zp[(size_t)rr * NC + ocol];
            v1 += beta * Zp[(size_t)rr * NC + ocol + 16];
        }
        Zn  [(size_t)rr * NC + ocol]           = v0;
        Zn  [(size_t)rr * NC + ocol + 16]      = v1;
        ZnH [(size_t)rr * NC + ocol]           = (_Float16)v0;
        ZnH [(size_t)rr * NC + ocol + 16]      = (_Float16)v1;
        ZnHT[(size_t)ocol * NN + rr]           = (_Float16)v0;
        ZnHT[(size_t)(ocol + 16) * NN + rr]    = (_Float16)v1;
    }
}

// ---------------------------------------------------------------------------
// out[r, :] = sum_k z_k[r, :] @ theta[k]   ==  Zcat[r, 0:256] @ ThetaT
// ThetaT staged to LDS via async copies: [64 cols][256 k] f16 (32 KB).
// ---------------------------------------------------------------------------
__global__ __launch_bounds__(256)
void cheb_epilogue(const _Float16* __restrict__ z0,
                   const _Float16* __restrict__ z1,
                   const _Float16* __restrict__ z2,
                   const _Float16* __restrict__ z3,
                   const _Float16* __restrict__ thT,   // [64][256]
                   float* __restrict__ out) {
    __shared__ __align__(128) _Float16 Ts[64][NK * NC];

    const int t = threadIdx.x;
    _Float16* tsf = &Ts[0][0];
    #pragma unroll
    for (int i = 0; i < 8; ++i)
        async_b128(tsf + (t * 8 + i) * 8, thT + (t * 8 + i) * 8);
    wait_asynccnt0();
    __syncthreads();

    const int w = t >> 5, l = t & 31;
    const int rowA = blockIdx.x * 128 + w * 16 + (l & 15);
    const int fkh  = (l >> 4) * 8;
    const _Float16* zs[4] = { z0, z1, z2, z3 };

    v8f acc[4] = { {}, {}, {}, {} };
    for (int kc = 0; kc < 8; ++kc) {                  // cat-K chunks of 32
        const _Float16* z = zs[kc >> 1];
        const int coff = (kc & 1) * 32;
        v16h a = load_frag_split(&z[(size_t)rowA * NC + coff + 2 * fkh],
                                 &z[(size_t)rowA * NC + coff + 2 * fkh + 8]);
        const int kg = kc * 32 + (l >> 4) * 16;
        #pragma unroll
        for (int n = 0; n < 4; ++n) {
            v16h bf = *(const v16h*)&Ts[n * 16 + (l & 15)][kg];
            acc[n] = __builtin_amdgcn_wmma_f32_16x16x32_f16(false, a, false, bf, (short)0, acc[n], false, false);
        }
    }

    const int orow = blockIdx.x * 128 + w * 16 + (l >> 4) * 8;
    const int ocol = l & 15;
    #pragma unroll
    for (int n = 0; n < 4; ++n)
        #pragma unroll
        for (int i = 0; i < 8; ++i)
            out[(size_t)(orow + i) * NC + n * 16 + ocol] = acc[n][i];
}

// ---------------------------------------------------------------------------
extern "C" void kernel_launch(void* const* d_in, const int* in_sizes, int n_in,
                              void* d_out, int out_size, void* d_ws, size_t ws_size,
                              hipStream_t stream) {
    const float* x  = (const float*)d_in[0];   // [8,1024,64]
    const float* L  = (const float*)d_in[1];   // [8,1024,1024]
    const float* th = (const float*)d_in[2];   // [4,64,64]
    float* out = (float*)d_out;                // [8,1024,64]

    const size_t nL = (size_t)NB * NN * NN;    // 8,388,608
    const size_t nZ = (size_t)NB * NN * NC;    // 524,288
    const size_t nT = (size_t)NK * NC * NC;    // 16,384

    char* p = (char*)d_ws;
    _Float16* Lh   = (_Float16*)p; p += nL * sizeof(_Float16);
    _Float16* xh   = (_Float16*)p; p += nZ * sizeof(_Float16);
    _Float16* xhT  = (_Float16*)p; p += nZ * sizeof(_Float16);
    _Float16* z1h  = (_Float16*)p; p += nZ * sizeof(_Float16);
    _Float16* z1hT = (_Float16*)p; p += nZ * sizeof(_Float16);
    _Float16* z2h  = (_Float16*)p; p += nZ * sizeof(_Float16);
    _Float16* z2hT = (_Float16*)p; p += nZ * sizeof(_Float16);
    _Float16* z3h  = (_Float16*)p; p += nZ * sizeof(_Float16);
    _Float16* z3hT = (_Float16*)p; p += nZ * sizeof(_Float16);
    float*    z1f  = (float*)p;    p += nZ * sizeof(float);
    float*    z2f  = (float*)p;    p += nZ * sizeof(float);
    float*    z3f  = (float*)p;    p += nZ * sizeof(float);
    _Float16* thT  = (_Float16*)p; p += nT * sizeof(_Float16);

    cvt_f32_to_f16<<<(int)(nL / 4 / 256), 256, 0, stream>>>(L, Lh, (int)(nL / 4));
    cvt_x_dual<<<(int)(nZ / 256), 256, 0, stream>>>(x, xh, xhT);
    pack_theta<<<(int)(nT / 256), 256, 0, stream>>>(th, thT);

    dim3 grid(NN / 64, NB);
    // z1 = L @ x
    cheb_gemmL<<<grid, 256, 0, stream>>>(Lh, xhT, x, z1f, z1h, z1hT, 1.0f, 0.0f);
    // z2 = 2 L z1 - x
    cheb_gemmL<<<grid, 256, 0, stream>>>(Lh, z1hT, x, z2f, z2h, z2hT, 2.0f, -1.0f);
    // z3 = 2 L z2 - z1
    cheb_gemmL<<<grid, 256, 0, stream>>>(Lh, z2hT, z1f, z3f, z3h, z3hT, 2.0f, -1.0f);
    // out = x@th0 + z1@th1 + z2@th2 + z3@th3
    cheb_epilogue<<<(NB * NN) / 128, 256, 0, stream>>>(xh, z1h, z2h, z3h, thT, out);
}
